// GNNAnomalyDetector_14783277433240
// MI455X (gfx1250) — matrix operand order repaired and checked by reading, not compiled, not run on hardware
//
#include <hip/hip_runtime.h>
#include <hip/hip_bf16.h>

// ---------------------------------------------------------------------------
// Problem constants
// ---------------------------------------------------------------------------
#define N_NODES 100000
#define E_MSG   3200000
#define E_SC    1000000
#define IN_DIM  12
#define HID     64
#define EMB     32
#define EDGE_F  10
#define ER      74      // 2*EMB + EDGE_F
#define ER_PAD  96      // K padded to 3x32 for WMMA

// Workspace layout (float offsets)
#define AGG1_OFF 0u            // [N_NODES*12]
#define CNT_OFF  1200000u      // [N_NODES]
#define AGG2_OFF 1300000u      // [N_NODES*64]
#define H_OFF    7700000u      // [N_NODES*64]
#define EMB_OFF  14100000u     // [N_NODES*32]
#define FRAG_BYTE_OFF 69200000u  // bf16 weight fragments (32B aligned)

// Fragment offsets (in bf16 halves); each 16x16-out x 32-K fragment = 512 halves
#define F_WS1 0
#define F_WN1 2048
#define F_WS2 4096
#define F_WN2 6144
#define F_E1  8192     // enc_W1: 3 ktiles x 4 ntiles
#define F_E2  14336    // enc_W2: 2 x 2
#define F_D1  16384    // dec_W1: 1 x 4
#define F_D2  18432    // dec_W2: 2 x 5  (edge-MLP frag set F_E1..end is contiguous: 15360 halves)

typedef __attribute__((ext_vector_type(16))) __bf16 v16bf;
typedef __attribute__((ext_vector_type(8)))  float  v8f;

// ---------------------------------------------------------------------------
// WMMA helpers
// ---------------------------------------------------------------------------
__device__ __forceinline__ v8f wmma_bf16(v16bf a, v16bf b, v8f c) {
  return __builtin_amdgcn_wmma_f32_16x16x32_bf16(
      /*neg_a=*/false, a, /*neg_b=*/false, b,
      /*c_mod=*/(short)0, c, /*reuse_a=*/false, /*reuse_b=*/false);
}

// B fragment from LDS-resident pack: 16 halves per lane, contiguous (32B/lane)
__device__ __forceinline__ v16bf ldb(const __bf16* p, int lane) {
  return *(const v16bf*)(p + lane * 16);
}

// Build 16x32 bf16 A fragment from a row-major f32 LDS tile.
// A layout: lanes 0-15 M=lane K in {0..7,16..23}; lanes 16-31 same M, K {8..15,24..31}
__device__ __forceinline__ v16bf a_frag_lds(const float* T, int stride, int m,
                                            int ko, int kbase) {
  v16bf a;
  const float* r = T + m * stride + kbase + ko;
#pragma unroll
  for (int hh = 0; hh < 16; hh += 2) {
    int K = ((hh >> 3) << 4) + (hh & 7);
    float2 v = *(const float2*)(r + K);
    a[hh] = (__bf16)v.x;
    a[hh + 1] = (__bf16)v.y;
  }
  return a;
}

// C fragment (lane = N, vgpr r = M row) -> f32 LDS tile, + bias (+ optional relu)
__device__ __forceinline__ void store_c(float* T, int stride, int nt, int lane,
                                        v8f acc, float bias, bool do_relu) {
  int n = nt * 16 + (lane & 15);
  int mo = (lane >> 4) * 8;
#pragma unroll
  for (int r = 0; r < 8; ++r) {
    float v = acc[r] + bias;
    if (do_relu) v = fmaxf(v, 0.f);
    T[(mo + r) * stride + n] = v;
  }
}

#define LDS_FENCE() asm volatile("s_wait_dscnt 0x0" ::: "memory")

// ---------------------------------------------------------------------------
// Utility kernels
// ---------------------------------------------------------------------------
__global__ void zero_f32(float* p, int n) {
  int i = blockIdx.x * blockDim.x + threadIdx.x;
  if (i < n) p[i] = 0.f;
}

// Pack W[K,N] (row-major f32) into bf16 B-fragment order:
// frag(kt,nt) base + lane*16 + h  holds  W[kt*32 + (lane>=16)*8 + (h>=8)*16 + h%8][nt*16 + lane%16]
__global__ void prep_frag(const float* W, int K, int N, int Nt, __bf16* dst, int total) {
  int t = blockIdx.x * blockDim.x + threadIdx.x;
  if (t >= total) return;
  int frag = t >> 9;
  int i = t & 511;
  int kt = frag / Nt, nt = frag % Nt;
  int L = i >> 4, h = i & 15;
  int k = kt * 32 + ((L >> 4) << 3) + ((h >> 3) << 4) + (h & 7);
  int n = nt * 16 + (L & 15);
  float v = (k < K && n < N) ? W[k * N + n] : 0.f;
  dst[t] = (__bf16)v;
}

// scatter x[src] into agg1[dst], count degrees
__global__ void scatter1(const float* __restrict__ x, const int* __restrict__ ei,
                         float* agg1, float* cnt, int E) {
  int e = blockIdx.x * blockDim.x + threadIdx.x;
  if (e >= E) return;
  int s = ei[e], d = ei[E + e];
  const float* xs = x + (size_t)s * IN_DIM;
  float* a = agg1 + (size_t)d * IN_DIM;
#pragma unroll
  for (int i = 0; i < IN_DIM; ++i) atomicAdd(a + i, xs[i]);
  atomicAdd(cnt + d, 1.f);
}

// scatter h[src] (64 feats) into agg2[dst]; thread per (edge, 16-feature chunk)
__global__ void scatter2(const float* __restrict__ hbuf, const int* __restrict__ ei,
                         float* agg2, int E) {
  int t = blockIdx.x * blockDim.x + threadIdx.x;
  if (t >= E * 4) return;
  int e = t >> 2, c = (t & 3) * 16;
  int s = ei[e], d = ei[E + e];
  const float* hs = hbuf + (size_t)s * HID + c;
  float* a = agg2 + (size_t)d * HID + c;
#pragma unroll
  for (int i = 0; i < 16; ++i) atomicAdd(a + i, hs[i]);
}

// cooperative global->LDS copy of weight fragments (uint4 granularity)
__device__ __forceinline__ void copy_frags_to_lds(__bf16* dst, const __bf16* src,
                                                  int n_u4, int tid, int nthreads) {
  const uint4* s = (const uint4*)src;
  uint4* d = (uint4*)dst;
  for (int i = tid; i < n_u4; i += nthreads) d[i] = s[i];
}

// ---------------------------------------------------------------------------
// SAGE layer 1: h = relu(LN(x@Ws1 + mean_agg@Wn1 + b1)); 12->64
// one 16-node tile per wave; weights LDS-resident
// ---------------------------------------------------------------------------
__global__ __launch_bounds__(128) void node_layer1(
    const float* __restrict__ x, const float* __restrict__ agg1,
    const float* __restrict__ cnt, const __bf16* __restrict__ frags,
    const float* __restrict__ b1, const float* __restrict__ g1,
    const float* __restrict__ be1, float* hbuf, int nTiles) {
  __shared__ float tile[4][16 * HID];
  __shared__ __bf16 wf[4096];  // Ws1(4 frags) + Wn1(4 frags)
  copy_frags_to_lds(wf, frags + F_WS1, 512, threadIdx.x, 128);
  __syncthreads();

  int wave = threadIdx.x >> 5, lane = threadIdx.x & 31;
  int t = blockIdx.x * 4 + wave;
  if (t >= nTiles) return;
  float* T = tile[wave];
  int m = lane & 15, ko = (lane >> 4) * 8;
  int row = t * 16 + m;
  const float* xr = x + (size_t)row * IN_DIM;
  const float* ar = agg1 + (size_t)row * IN_DIM;
  float rc = 1.f / fmaxf(cnt[row], 1.f);

  v16bf ax, ag;
#pragma unroll
  for (int hh = 0; hh < 16; hh += 2) {
    int K = ko + ((hh >> 3) << 4) + (hh & 7);
    float x0 = 0.f, x1 = 0.f, a0 = 0.f, a1 = 0.f;
    if (K <= IN_DIM - 2) {
      float2 u = *(const float2*)(xr + K);
      float2 w = *(const float2*)(ar + K);
      x0 = u.x; x1 = u.y; a0 = w.x * rc; a1 = w.y * rc;
    }
    ax[hh] = (__bf16)x0; ax[hh + 1] = (__bf16)x1;
    ag[hh] = (__bf16)a0; ag[hh + 1] = (__bf16)a1;
  }

  v8f acc[4];
#pragma unroll
  for (int nt = 0; nt < 4; ++nt) {
    v8f c = {};
    c = wmma_bf16(ax, ldb(wf + nt * 512, lane), c);          // Ws1
    c = wmma_bf16(ag, ldb(wf + 2048 + nt * 512, lane), c);   // Wn1
    acc[nt] = c;
  }
#pragma unroll
  for (int nt = 0; nt < 4; ++nt)
    store_c(T, HID, nt, lane, acc[nt], b1[nt * 16 + m], false);
  LDS_FENCE();

  if (lane < 16) {
    const float* rrow = T + lane * HID;
    float s = 0.f, s2 = 0.f;
#pragma unroll
    for (int i = 0; i < HID; ++i) { float v = rrow[i]; s += v; s2 += v * v; }
    float mu = s * (1.f / HID);
    float var = s2 * (1.f / HID) - mu * mu;
    float rstd = rsqrtf(var + 1e-5f);
    float* orow = hbuf + (size_t)(t * 16 + lane) * HID;
#pragma unroll
    for (int i = 0; i < HID; ++i) {
      float v = (rrow[i] - mu) * rstd * g1[i] + be1[i];
      orow[i] = fmaxf(v, 0.f);
    }
  }
}

// ---------------------------------------------------------------------------
// SAGE layer 2: emb = relu(LN(h@Ws2 + mean_agg2@Wn2 + b2)); 64->32
// ---------------------------------------------------------------------------
__global__ __launch_bounds__(128) void node_layer2(
    const float* __restrict__ hbuf, const float* __restrict__ agg2,
    const float* __restrict__ cnt, const __bf16* __restrict__ frags,
    const float* __restrict__ b2, const float* __restrict__ g2,
    const float* __restrict__ be2, float* embbuf, int nTiles) {
  __shared__ float tile[4][16 * EMB];
  __shared__ __bf16 wf[4096];  // Ws2(4 frags) + Wn2(4 frags)
  copy_frags_to_lds(wf, frags + F_WS2, 512, threadIdx.x, 128);
  __syncthreads();

  int wave = threadIdx.x >> 5, lane = threadIdx.x & 31;
  int t = blockIdx.x * 4 + wave;
  if (t >= nTiles) return;
  float* T = tile[wave];
  int m = lane & 15, ko = (lane >> 4) * 8;
  int row = t * 16 + m;
  const float* hr = hbuf + (size_t)row * HID;
  const float* ar = agg2 + (size_t)row * HID;
  float rc = 1.f / fmaxf(cnt[row], 1.f);

  v16bf ah[2], am[2];
#pragma unroll
  for (int kt = 0; kt < 2; ++kt) {
#pragma unroll
    for (int hh = 0; hh < 16; hh += 2) {
      int K = kt * 32 + ko + ((hh >> 3) << 4) + (hh & 7);
      float2 u = *(const float2*)(hr + K);
      float2 w = *(const float2*)(ar + K);
      ah[kt][hh] = (__bf16)u.x; ah[kt][hh + 1] = (__bf16)u.y;
      am[kt][hh] = (__bf16)(w.x * rc); am[kt][hh + 1] = (__bf16)(w.y * rc);
    }
  }

  v8f acc[2];
#pragma unroll
  for (int nt = 0; nt < 2; ++nt) {
    v8f c = {};
    c = wmma_bf16(ah[0], ldb(wf + (0 * 2 + nt) * 512, lane), c);
    c = wmma_bf16(ah[1], ldb(wf + (1 * 2 + nt) * 512, lane), c);
    c = wmma_bf16(am[0], ldb(wf + 2048 + (0 * 2 + nt) * 512, lane), c);
    c = wmma_bf16(am[1], ldb(wf + 2048 + (1 * 2 + nt) * 512, lane), c);
    acc[nt] = c;
  }
#pragma unroll
  for (int nt = 0; nt < 2; ++nt)
    store_c(T, EMB, nt, lane, acc[nt], b2[nt * 16 + m], false);
  LDS_FENCE();

  if (lane < 16) {
    const float* rrow = T + lane * EMB;
    float s = 0.f, s2 = 0.f;
#pragma unroll
    for (int i = 0; i < EMB; ++i) { float v = rrow[i]; s += v; s2 += v * v; }
    float mu = s * (1.f / EMB);
    float var = s2 * (1.f / EMB) - mu * mu;
    float rstd = rsqrtf(var + 1e-5f);
    float* orow = embbuf + (size_t)(t * 16 + lane) * EMB;
#pragma unroll
    for (int i = 0; i < EMB; ++i) {
      float v = (rrow[i] - mu) * rstd * g2[i] + be2[i];
      orow[i] = fmaxf(v, 0.f);
    }
  }
}

// ---------------------------------------------------------------------------
// Edge autoencoder: gather edge_rep (74), write it out, run 74->64->32->64->74
// 8 waves/block, one 16-edge tile per wave; all MLP weights LDS-resident
// Local LDS frag offsets (halves): E1=0, E2=6144, D1=8192, D2=10240
// ---------------------------------------------------------------------------
__global__ __launch_bounds__(256) void edge_mlp(
    const float* __restrict__ emb, const float* __restrict__ eattr,
    const int* __restrict__ srcn, const int* __restrict__ dstn,
    const __bf16* __restrict__ frags,
    const float* __restrict__ enc_b1, const float* __restrict__ enc_b2,
    const float* __restrict__ dec_b1, const float* __restrict__ dec_b2,
    float* out_recon, float* out_rep, int nTiles) {
  __shared__ float tile[8][16 * ER_PAD];   // 48 KB
  __shared__ __bf16 wf[15360];             // 30 frags = 30 KB (E1|E2|D1|D2 contiguous)
  copy_frags_to_lds(wf, frags + F_E1, 1920, threadIdx.x, 256);
  __syncthreads();

  int wave = threadIdx.x >> 5, lane = threadIdx.x & 31;
  int t = blockIdx.x * 8 + wave;
  if (t >= nTiles) return;
  float* T = tile[wave];
  int m = lane & 15, ko = (lane >> 4) * 8;
  int e = t * 16 + m;
  float* orep = out_rep + (size_t)e * ER;

  // ---- gather edge_rep into LDS tile (and to output) ----
  if (lane < 16) {
    const float* es = emb + (size_t)srcn[e] * EMB;
#pragma unroll
    for (int i = 0; i < EMB; i += 2) {
      float2 v = *(const float2*)(es + i);
      *(float2*)(T + m * ER_PAD + i) = v;
      *(float2*)(orep + i) = v;
    }
  } else {
    const float* ed = emb + (size_t)dstn[e] * EMB;
#pragma unroll
    for (int i = 0; i < EMB; i += 2) {
      float2 v = *(const float2*)(ed + i);
      *(float2*)(T + m * ER_PAD + EMB + i) = v;
      *(float2*)(orep + EMB + i) = v;
    }
    const float* ea = eattr + (size_t)e * EDGE_F;
#pragma unroll
    for (int i = 0; i < EDGE_F; i += 2) {
      float2 v = *(const float2*)(ea + i);
      *(float2*)(T + m * ER_PAD + 2 * EMB + i) = v;
      *(float2*)(orep + 2 * EMB + i) = v;
    }
#pragma unroll
    for (int i = ER; i < ER_PAD; i += 2)
      *(float2*)(T + m * ER_PAD + i) = make_float2(0.f, 0.f);
  }
  LDS_FENCE();

  // ---- encoder L1: 96(K) -> 64 ----
  v16bf a0 = a_frag_lds(T, ER_PAD, m, ko, 0);
  v16bf a1 = a_frag_lds(T, ER_PAD, m, ko, 32);
  v16bf a2 = a_frag_lds(T, ER_PAD, m, ko, 64);
  v8f acc4[4];
#pragma unroll
  for (int nt = 0; nt < 4; ++nt) {
    v8f c = {};
    c = wmma_bf16(a0, ldb(wf + (0 * 4 + nt) * 512, lane), c);
    c = wmma_bf16(a1, ldb(wf + (1 * 4 + nt) * 512, lane), c);
    c = wmma_bf16(a2, ldb(wf + (2 * 4 + nt) * 512, lane), c);
    acc4[nt] = c;
  }
#pragma unroll
  for (int nt = 0; nt < 4; ++nt)
    store_c(T, ER_PAD, nt, lane, acc4[nt], enc_b1[nt * 16 + m], true);
  LDS_FENCE();

  // ---- encoder L2: 64 -> 32 (latent) ----
  a0 = a_frag_lds(T, ER_PAD, m, ko, 0);
  a1 = a_frag_lds(T, ER_PAD, m, ko, 32);
  v8f acc2[2];
#pragma unroll
  for (int nt = 0; nt < 2; ++nt) {
    v8f c = {};
    c = wmma_bf16(a0, ldb(wf + 6144 + (0 * 2 + nt) * 512, lane), c);
    c = wmma_bf16(a1, ldb(wf + 6144 + (1 * 2 + nt) * 512, lane), c);
    acc2[nt] = c;
  }
#pragma unroll
  for (int nt = 0; nt < 2; ++nt)
    store_c(T, ER_PAD, nt, lane, acc2[nt], enc_b2[nt * 16 + m], true);
  LDS_FENCE();

  // ---- decoder L1: 32 -> 64 ----
  a0 = a_frag_lds(T, ER_PAD, m, ko, 0);
#pragma unroll
  for (int nt = 0; nt < 4; ++nt) {
    v8f c = {};
    acc4[nt] = wmma_bf16(a0, ldb(wf + 8192 + nt * 512, lane), c);
  }
#pragma unroll
  for (int nt = 0; nt < 4; ++nt)
    store_c(T, ER_PAD, nt, lane, acc4[nt], dec_b1[nt * 16 + m], true);
  LDS_FENCE();

  // ---- decoder L2: 64 -> 74 (padded 80) ----
  a0 = a_frag_lds(T, ER_PAD, m, ko, 0);
  a1 = a_frag_lds(T, ER_PAD, m, ko, 32);
  v8f acc5[5];
#pragma unroll
  for (int nt = 0; nt < 5; ++nt) {
    v8f c = {};
    c = wmma_bf16(a0, ldb(wf + 10240 + (0 * 5 + nt) * 512, lane), c);
    c = wmma_bf16(a1, ldb(wf + 10240 + (1 * 5 + nt) * 512, lane), c);
    acc5[nt] = c;
  }
  int mo = (lane >> 4) * 8;
#pragma unroll
  for (int nt = 0; nt < 5; ++nt) {
    int n = nt * 16 + m;
    if (n < ER) {
      float bias = dec_b2[n];
#pragma unroll
      for (int r = 0; r < 8; ++r) {
        int row = t * 16 + mo + r;
        out_recon[(size_t)row * ER + n] = acc5[nt][r] + bias;
      }
    }
  }
}

// ---------------------------------------------------------------------------
// Host entry
// ---------------------------------------------------------------------------
extern "C" void kernel_launch(void* const* d_in, const int* in_sizes, int n_in,
                              void* d_out, int out_size, void* d_ws, size_t ws_size,
                              hipStream_t stream) {
  const float* node_feats = (const float*)d_in[0];
  const int*   edge_index = (const int*)d_in[1];
  const float* edge_attr  = (const float*)d_in[2];
  const int*   src_nodes  = (const int*)d_in[3];
  const int*   dst_nodes  = (const int*)d_in[4];
  const float* Ws1 = (const float*)d_in[5];
  const float* Wn1 = (const float*)d_in[6];
  const float* b1  = (const float*)d_in[7];
  const float* g1  = (const float*)d_in[8];
  const float* be1 = (const float*)d_in[9];
  const float* Ws2 = (const float*)d_in[10];
  const float* Wn2 = (const float*)d_in[11];
  const float* b2  = (const float*)d_in[12];
  const float* g2  = (const float*)d_in[13];
  const float* be2 = (const float*)d_in[14];
  const float* enc_W1 = (const float*)d_in[15];
  const float* enc_b1 = (const float*)d_in[16];
  const float* enc_W2 = (const float*)d_in[17];
  const float* enc_b2 = (const float*)d_in[18];
  const float* dec_W1 = (const float*)d_in[19];
  const float* dec_b1 = (const float*)d_in[20];
  const float* dec_W2 = (const float*)d_in[21];
  const float* dec_b2 = (const float*)d_in[22];

  float* ws   = (float*)d_ws;
  float* agg1 = ws + AGG1_OFF;
  float* cnt  = ws + CNT_OFF;
  float* agg2 = ws + AGG2_OFF;
  float* hbuf = ws + H_OFF;
  float* embb = ws + EMB_OFF;
  __bf16* frags = (__bf16*)((char*)d_ws + FRAG_BYTE_OFF);

  float* out_recon = (float*)d_out;
  float* out_rep   = (float*)d_out + (size_t)E_SC * ER;

  // 1) zero aggregation buffers (agg1 + cnt + agg2)
  {
    int n = (int)(AGG2_OFF + (size_t)N_NODES * HID);  // 7,700,000 floats
    zero_f32<<<(n + 255) / 256, 256, 0, stream>>>(ws, n);
  }

  // 2) pack weights into bf16 B-fragments
  prep_frag<<<(2048 + 255) / 256, 256, 0, stream>>>(Ws1, 12, 64, 4, frags + F_WS1, 2048);
  prep_frag<<<(2048 + 255) / 256, 256, 0, stream>>>(Wn1, 12, 64, 4, frags + F_WN1, 2048);
  prep_frag<<<(2048 + 255) / 256, 256, 0, stream>>>(Ws2, 64, 32, 2, frags + F_WS2, 2048);
  prep_frag<<<(2048 + 255) / 256, 256, 0, stream>>>(Wn2, 64, 32, 2, frags + F_WN2, 2048);
  prep_frag<<<(6144 + 255) / 256, 256, 0, stream>>>(enc_W1, 74, 64, 4, frags + F_E1, 6144);
  prep_frag<<<(2048 + 255) / 256, 256, 0, stream>>>(enc_W2, 64, 32, 2, frags + F_E2, 2048);
  prep_frag<<<(2048 + 255) / 256, 256, 0, stream>>>(dec_W1, 32, 64, 4, frags + F_D1, 2048);
  prep_frag<<<(5120 + 255) / 256, 256, 0, stream>>>(dec_W2, 64, 74, 5, frags + F_D2, 5120);

  // 3) SAGE layer 1
  scatter1<<<(E_MSG + 255) / 256, 256, 0, stream>>>(node_feats, edge_index, agg1, cnt, E_MSG);
  {
    int nTiles = N_NODES / 16;  // 6250
    node_layer1<<<(nTiles + 3) / 4, 128, 0, stream>>>(node_feats, agg1, cnt, frags,
                                                      b1, g1, be1, hbuf, nTiles);
  }

  // 4) SAGE layer 2
  scatter2<<<(E_MSG * 4 + 255) / 256, 256, 0, stream>>>(hbuf, edge_index, agg2, E_MSG);
  {
    int nTiles = N_NODES / 16;
    node_layer2<<<(nTiles + 3) / 4, 128, 0, stream>>>(hbuf, agg2, cnt, frags,
                                                      b2, g2, be2, embb, nTiles);
  }

  // 5) edge autoencoder (also emits edge_rep output)
  {
    int nTiles = E_SC / 16;  // 62500
    edge_mlp<<<(nTiles + 7) / 8, 256, 0, stream>>>(embb, edge_attr, src_nodes, dst_nodes,
                                                   frags, enc_b1, enc_b2, dec_b1, dec_b2,
                                                   out_recon, out_rep, nTiles);
  }
}
